// SimpleESN_27453430956033
// MI455X (gfx1250) — matrix operand order repaired and checked by reading, not compiled, not run on hardware
//
#include <hip/hip_runtime.h>
#include <hip/hip_bf16.h>

// Problem constants (match reference)
#define B_ 16
#define T_ 2048
#define F_ 128
#define U_ 2048

#define NTILES   (U_ / 16)   // 128 output column tiles, one wave each
#define KBLK_U   (U_ / 32)   // 64 K-blocks for recurrent term
#define KBLK_F   (F_ / 32)   // 4 K-blocks for input projection
#define FRAG_EL  512         // elements per fragment block: 32 lanes * 16 bf16

typedef __bf16 bf16_t;
typedef bf16_t v16bf __attribute__((ext_vector_type(16)));
typedef float  v8f   __attribute__((ext_vector_type(8)));
typedef unsigned int u32x4 __attribute__((ext_vector_type(4)));

union Frag {
  v16bf v;
  u32x4 q[2];   // 32 bytes = 16 bf16 (one lane's fragment slice)
};

__device__ __forceinline__ unsigned short f32_to_bf16_bits(float f) {
  unsigned u = __float_as_uint(f);
  unsigned r = u + 0x7FFFu + ((u >> 16) & 1u);   // round-to-nearest-even
  return (unsigned short)(r >> 16);
}

// ---------------------------------------------------------------------------
// Fragment layouts (match ISA 16-bit WMMA VGPR striping):
//  A-frag block (16x32, K-block kb): lane l (half=l>>4, row=l&15) holds
//    elems 0-7  = A[row][kb*32 + half*8 + 0..7]
//    elems 8-15 = A[row][kb*32 + 16 + half*8 + 0..7]
//  B-frag block (32x16): lane l holds B[kb*32 + k][n0 + (l&15)], k per elem as in A-frag
//  Storage: frag-major, lane-contiguous: idx = (block*32 + lane)*16 + elem
//  -> every hot-loop load is base + lane*32B, fully coalesced.
// ---------------------------------------------------------------------------

// W [U][U] f32 -> Wopt[ntile][kb][lane][16] bf16
__global__ void reorder_W(const float* __restrict__ W,
                          unsigned short* __restrict__ Wopt) {
  int i = blockIdx.x * blockDim.x + threadIdx.x;
  int stride = gridDim.x * blockDim.x;
  const int n = NTILES * KBLK_U * FRAG_EL;      // 4M
  for (; i < n; i += stride) {
    int e  = i & 15;
    int l  = (i >> 4) & 31;
    int kb = (i >> 9) & (KBLK_U - 1);
    int nt = i >> 15;
    Wopt[i] = f32_to_bf16_bits(W[(kb * 32 + ((e & 8) ? (16 + (l >> 4) * 8 + (e & 7)) : ((l >> 4) * 8 + e))) * U_ + nt * 16 + (l & 15)]);
  }
}

// kernel [F][U] f32 -> Kopt[ntile][kb][lane][16] bf16
__global__ void reorder_K(const float* __restrict__ K,
                          unsigned short* __restrict__ Kopt) {
  int i = blockIdx.x * blockDim.x + threadIdx.x;
  int stride = gridDim.x * blockDim.x;
  const int n = NTILES * KBLK_F * FRAG_EL;      // 256K
  for (; i < n; i += stride) {
    int e  = i & 15;
    int l  = (i >> 4) & 31;
    int kb = (i >> 9) & (KBLK_F - 1);
    int nt = i >> 11;
    Kopt[i] = f32_to_bf16_bits(K[(kb * 32 + ((e & 8) ? (16 + (l >> 4) * 8 + (e & 7)) : ((l >> 4) * 8 + e))) * U_ + nt * 16 + (l & 15)]);
  }
}

// x [B][T][F] f32 -> Xopt[t][kb][lane][16] bf16 (A-fragment layout per step)
__global__ void reorder_X(const float* __restrict__ X,
                          unsigned short* __restrict__ Xopt) {
  int i = blockIdx.x * blockDim.x + threadIdx.x;
  int stride = gridDim.x * blockDim.x;
  const int n = T_ * KBLK_F * FRAG_EL;          // 4M
  for (; i < n; i += stride) {
    int e    = i & 15;
    int l    = (i >> 4) & 31;
    int kb   = (i >> 9) & (KBLK_F - 1);
    int t    = i >> 11;
    int b    = l & 15;
    int half = l >> 4;
    int f    = kb * 32 + ((e & 8) ? (16 + half * 8 + (e & 7))
                                  : (half * 8 + e));
    Xopt[i] = f32_to_bf16_bits(X[(b * T_ + t) * F_ + f]);
  }
}

// ---------------------------------------------------------------------------
// Persistent ESN scan. One wave owns one 16-column tile of U (128 waves).
// h state lives in A-fragment layout (64 blocks * 1KB = 64KB bf16), double-
// buffered. All hot-loop loads are contiguous base + lane*32B streams.
// ---------------------------------------------------------------------------
__global__ void __launch_bounds__(256, 1)
esn_scan(const unsigned short* __restrict__ Wopt,   // [NTILES][KBLK_U][512]
         const unsigned short* __restrict__ Kopt,   // [NTILES][KBLK_F][512]
         const unsigned short* __restrict__ Xopt,   // [T][KBLK_F][512]
         const float* __restrict__ bias,            // [U] f32
         unsigned short* __restrict__ h0buf,        // [KBLK_U][512] bf16
         unsigned short* __restrict__ h1buf,        // [KBLK_U][512] bf16
         float* __restrict__ hf32,                  // [B][U] f32 final state
         int* __restrict__ counter)                 // grid barrier
{
  const int lane  = threadIdx.x & 31;
  const int wave  = threadIdx.x >> 5;
  const int gwave = blockIdx.x * (blockDim.x >> 5) + wave;  // = ntile
  const int n0    = gwave * 16;
  const int half  = lane >> 4;           // C/D: rows r+8*half
  const int mlo   = lane & 15;           // C/D: column n0+mlo
  const int col   = n0 + mlo;
  const float bval = bias[col];
  const int nblocks = gridDim.x;

  // Output scatter coords into A-fragment layout of h_next (col -> K dim)
  const int kbc   = col >> 5;                       // dest K-block
  const int koff  = col & 31;
  const int ee    = (koff & 7) + ((koff & 16) >> 1);  // fragment elem 0..15
  const int halfp = (koff >> 3) & 1;                  // dest lane half
  const int doff  = kbc * FRAG_EL + halfp * 256 + ee; // + m*16 per row

  const unsigned short* wbase = Wopt + (size_t)gwave * (KBLK_U * FRAG_EL);
  const unsigned short* kbase = Kopt + (size_t)gwave * (KBLK_F * FRAG_EL);
  const int loff = lane * 16;            // this lane's 32B within a block

  for (int t = 0; t < T_; ++t) {
    const unsigned short* hsrc = (t & 1) ? h1buf : h0buf;
    unsigned short*       hdst = (t & 1) ? h0buf : h1buf;

    v8f acc = {};

    // ---- recurrent term: 64 K-blocks, fully sequential 1KB streams --------
    for (int kb = 0; kb < KBLK_U; ++kb) {
      Frag a, b;
      const unsigned short* ap = hsrc  + kb * FRAG_EL + loff;
      const unsigned short* bp = wbase + kb * FRAG_EL + loff;
      a.q[0] = *(const u32x4*)(ap);
      a.q[1] = *(const u32x4*)(ap + 8);
      b.q[0] = *(const u32x4*)(bp);
      b.q[1] = *(const u32x4*)(bp + 8);
      __builtin_prefetch(bp + 2 * FRAG_EL, 0, 1);   // stream 2KB ahead of W
      acc = __builtin_amdgcn_wmma_f32_16x16x32_bf16(
          /*neg_a=*/false, a.v, /*neg_b=*/false, b.v,
          /*c_mod=*/(short)0, acc, /*reuse_a=*/false, /*reuse_b=*/false);
    }

    // ---- input projection: x[:,t,:] @ kernel (4 K-blocks) -----------------
    const unsigned short* xbase = Xopt + (size_t)t * (KBLK_F * FRAG_EL);
#pragma unroll
    for (int kb = 0; kb < KBLK_F; ++kb) {
      Frag a, b;
      const unsigned short* ap = xbase + kb * FRAG_EL + loff;
      const unsigned short* bp = kbase + kb * FRAG_EL + loff;
      a.q[0] = *(const u32x4*)(ap);
      a.q[1] = *(const u32x4*)(ap + 8);
      b.q[0] = *(const u32x4*)(bp);
      b.q[1] = *(const u32x4*)(bp + 8);
      acc = __builtin_amdgcn_wmma_f32_16x16x32_bf16(
          false, a.v, false, b.v, (short)0, acc, false, false);
    }

    // ---- activation + state write (into A-fragment layout) ----------------
#pragma unroll
    for (int r = 0; r < 8; ++r) {
      int m = r + half * 8;
      float v = tanhf(acc[r] + bval);          // LEAKY == 1.0
      hdst[doff + m * 16] = f32_to_bf16_bits(v);
      if (t == T_ - 1) hf32[m * U_ + col] = v;
    }

    // ---- grid-wide barrier: release, arrive, relaxed spin, acquire --------
    __threadfence();
    __syncthreads();
    if (threadIdx.x == 0) atomicAdd(counter, 1);
    const int expected = nblocks * (t + 1);
    while (__hip_atomic_load(counter, __ATOMIC_RELAXED,
                             __HIP_MEMORY_SCOPE_AGENT) < expected) {
      __builtin_amdgcn_s_sleep(1);
    }
    __threadfence();
  }
}

// ---------------------------------------------------------------------------
// Readout: out[b] = sum_u h[b,u] * w_out[u] + b_out[0]   (P == 1)
// ---------------------------------------------------------------------------
__global__ void readout(const float* __restrict__ h,
                        const float* __restrict__ w_out,
                        const float* __restrict__ b_out,
                        float* __restrict__ out) {
  __shared__ float red[256];
  const int b = blockIdx.x;
  float s = 0.0f;
  for (int i = threadIdx.x; i < U_; i += blockDim.x)
    s += h[b * U_ + i] * w_out[i];
  red[threadIdx.x] = s;
  __syncthreads();
  for (int off = 128; off > 0; off >>= 1) {
    if (threadIdx.x < off) red[threadIdx.x] += red[threadIdx.x + off];
    __syncthreads();
  }
  if (threadIdx.x == 0) out[b] = red[0] + b_out[0];
}

// ---------------------------------------------------------------------------
extern "C" void kernel_launch(void* const* d_in, const int* in_sizes, int n_in,
                              void* d_out, int out_size, void* d_ws, size_t ws_size,
                              hipStream_t stream) {
  const float* x     = (const float*)d_in[0];   // [B,T,F]
  const float* kern  = (const float*)d_in[1];   // [F,U]
  const float* W     = (const float*)d_in[2];   // [U,U]
  const float* bias  = (const float*)d_in[3];   // [U]
  const float* w_out = (const float*)d_in[4];   // [U,P], P=1
  const float* b_out = (const float*)d_in[5];   // [P]

  // Workspace carve-up (256B aligned), total ~17.6 MB
  char* ws = (char*)d_ws;
  size_t off = 0;
  auto carve = [&](size_t bytes) -> void* {
    void* p = ws + off;
    off = (off + bytes + 255) & ~(size_t)255;
    return p;
  };
  unsigned short* Wopt  = (unsigned short*)carve((size_t)NTILES * KBLK_U * FRAG_EL * 2);
  unsigned short* Kopt  = (unsigned short*)carve((size_t)NTILES * KBLK_F * FRAG_EL * 2);
  unsigned short* Xopt  = (unsigned short*)carve((size_t)T_ * KBLK_F * FRAG_EL * 2);
  unsigned short* h0buf = (unsigned short*)carve((size_t)KBLK_U * FRAG_EL * 2);
  unsigned short* h1buf = (unsigned short*)carve((size_t)KBLK_U * FRAG_EL * 2);
  float*          hf32  = (float*)carve((size_t)B_ * U_ * 4);
  int*            cnt   = (int*)carve(256);
  (void)ws_size; (void)n_in; (void)in_sizes; (void)out_size;

  // One-time fragment-major reorders (off the scan's critical path)
  reorder_W<<<2048, 256, 0, stream>>>(W,    Wopt);
  reorder_K<<<256,  256, 0, stream>>>(kern, Kopt);
  reorder_X<<<2048, 256, 0, stream>>>(x,    Xopt);

  // h0 = 0, barrier counter = 0 (graph replays must be identical)
  hipMemsetAsync(h0buf, 0, (size_t)KBLK_U * FRAG_EL * 2, stream);
  hipMemsetAsync(cnt,   0, 256, stream);

  // Persistent scan: 16 blocks x 256 threads = 128 waves = U/16 column tiles
  esn_scan<<<16, 256, 0, stream>>>(Wopt, Kopt, Xopt, bias, h0buf, h1buf, hf32, cnt);

  // Dense readout on final f32 state
  readout<<<B_, 256, 0, stream>>>(hf32, w_out, b_out, (float*)d_out);
}
